// cross_attention_36215164239898
// MI455X (gfx1250) — compile-verified, matrix operand order checked by reading, not run on hardware
//
#include <hip/hip_runtime.h>

// ---------------------------------------------------------------------------
// CDNA5 (gfx1250) wave32 WMMA types
// ---------------------------------------------------------------------------
typedef __attribute__((ext_vector_type(16))) __bf16 v16bf;
typedef __attribute__((ext_vector_type(8)))  float  v8f;
typedef __attribute__((ext_vector_type(4)))  unsigned int v4u;
typedef __attribute__((ext_vector_type(8)))  int v8i_t;
typedef __attribute__((ext_vector_type(4)))  int v4i_t;

union Frag16 { v16bf v; unsigned int u[8]; };

// Build a 16x32 (MxK) bf16 A-fragment (B-fragment uses the same per-lane
// layout with N in the role of M) from a row pointer with contiguous K.
//   lanes  0..15 : K in {0..7, 16..23}
//   lanes 16..31 : K in {8..15, 24..31}
__device__ __forceinline__ v16bf frag_from_row(const __bf16* row, bool hi) {
  const unsigned int* p = (const unsigned int*)row; // bf16 pairs
  const int o = hi ? 4 : 0;
  Frag16 f;
#pragma unroll
  for (int j = 0; j < 4; ++j) f.u[j] = p[o + j];         // K 0..7  / 8..15
#pragma unroll
  for (int j = 0; j < 4; ++j) f.u[4 + j] = p[o + 8 + j]; // K 16..23 / 24..31
  return f.v;
}

__device__ __forceinline__ v8f wmma_bf16(v16bf a, v16bf b, v8f c) {
  return __builtin_amdgcn_wmma_f32_16x16x32_bf16(false, a, false, b,
                                                 (short)0, c, false, false);
}

// ---------------------------------------------------------------------------
// Pure-VALU 16-lane butterfly reductions via DPP (no ds_bpermute, no dscnt
// waits).  xor1/xor2 = quad_perm, xor-in-8 = row_half_mirror (0x141),
// xor-in-16 = row_mirror (0x140).  Exact butterfly for sum; valid for max.
// ---------------------------------------------------------------------------
template <int CTRL>
__device__ __forceinline__ float dppf(float x) {
  return __builtin_bit_cast(
      float, __builtin_amdgcn_mov_dpp(__builtin_bit_cast(int, x), CTRL, 0xf,
                                      0xf, true));
}
__device__ __forceinline__ float redmax16(float v) {
  v = fmaxf(v, dppf<0xB1>(v));    // quad_perm [1,0,3,2]  (xor 1)
  v = fmaxf(v, dppf<0x4E>(v));    // quad_perm [2,3,0,1]  (xor 2)
  v = fmaxf(v, dppf<0x141>(v));   // row_half_mirror      (xor 7 -> pairs quads)
  v = fmaxf(v, dppf<0x140>(v));   // row_mirror           (xor 15 -> halves)
  return v;
}
__device__ __forceinline__ float redsum16(float v) {
  v += dppf<0xB1>(v);
  v += dppf<0x4E>(v);
  v += dppf<0x141>(v);
  v += dppf<0x140>(v);
  return v;
}

// ---------------------------------------------------------------------------
// Problem constants
// ---------------------------------------------------------------------------
#define BATCH 8
#define INCH  64
#define MIDCH 32
#define HH    128
#define WW    128
#define PH    64
#define PW    64
#define NTOK  4096

// ---------------------------------------------------------------------------
// Tensor Data Mover: build a D# and issue tensor_load_to_lds (clang-23 /
// therock 6-arg builtin).  2-D tile of 2-byte elements; dims/strides in
// elements, global address in bytes; lds_addr = low 32 bits of the generic
// pointer (flat->LDS aperture mapping keeps the byte offset in addr[31:0]).
// ---------------------------------------------------------------------------
#if __has_builtin(__builtin_amdgcn_tensor_load_to_lds)
#define USE_TDM 1
__device__ __forceinline__ void tdm_load_2d_bf16(void* lds_ptr,
                                                 const void* gptr,
                                                 unsigned tile_d0,
                                                 unsigned tile_d1,
                                                 unsigned tdim0, unsigned tdim1,
                                                 unsigned long long stride0) {
  const unsigned long long ga = (unsigned long long)gptr;
  const unsigned lds_off = (unsigned)(unsigned long long)lds_ptr;
  v4u g0;
  g0[0] = 1u;                                      // count=1, user mode
  g0[1] = lds_off;                                 // lds_addr
  g0[2] = (unsigned)ga;                            // global_addr[31:0]
  g0[3] = ((unsigned)(ga >> 32) & 0x1FFFFFFu) | (2u << 30);  // type=2
  v8i_t g1;
  g1[0] = (int)(1u << 16);                         // data_size=1 (2 bytes)
  g1[1] = (int)((tdim0 & 0xFFFFu) << 16);          // tensor_dim0[15:0]
  g1[2] = (int)((tdim0 >> 16) | ((tdim1 & 0xFFFFu) << 16));
  g1[3] = (int)((tdim1 >> 16) | (tile_d0 << 16));  // tile_dim0
  g1[4] = (int)(tile_d1 & 0xFFFFu);                // tile_dim1 (tile_dim2=0)
  g1[5] = (int)(stride0 & 0xFFFFFFFFu);            // tensor_dim0_stride lo
  g1[6] = (int)((stride0 >> 32) & 0xFFFFu);        // stride hi; dim1_stride=0
  g1[7] = 0;
  const v4i_t z4 = {0, 0, 0, 0};
  const v8i_t z8 = {0, 0, 0, 0, 0, 0, 0, 0};
  __builtin_amdgcn_tensor_load_to_lds(g0, g1, z4, z4, z8, 0);
}
#else
#define USE_TDM 0
#endif

// ---------------------------------------------------------------------------
// Kernel 1: fused 2x2 avg-pool + three 1x1-conv projections.
// Emits bf16 theta[b][n][32], phi[b][n][32], gT[b][32][n]
// grid = (PH, BATCH), block = 256
// ---------------------------------------------------------------------------
__global__ void prep_kernel(const float* __restrict__ primary,
                            const float* __restrict__ cross,
                            const float* __restrict__ w_theta,
                            const float* __restrict__ w_phi,
                            const float* __restrict__ w_g,
                            __bf16* __restrict__ thetaB,
                            __bf16* __restrict__ phiB,
                            __bf16* __restrict__ gT) {
  const int y = blockIdx.x;
  const int b = blockIdx.y;
  const int t = threadIdx.x;

  __shared__ float pp[INCH * PW];
  __shared__ float pc[INCH * PW];
  __shared__ float wt[MIDCH * INCH];
  __shared__ float wp[MIDCH * INCH];
  __shared__ float wg[MIDCH * INCH];

  for (int i = t; i < MIDCH * INCH; i += 256) {
    wt[i] = w_theta[i]; wp[i] = w_phi[i]; wg[i] = w_g[i];
  }

  const float* pb = primary + (size_t)b * INCH * HH * WW;
  const float* cb = cross   + (size_t)b * INCH * HH * WW;
  for (int e = t; e < INCH * PW; e += 256) {
    const int c = e >> 6, x = e & 63;
    const size_t base = (size_t)c * HH * WW + (size_t)(2 * y) * WW + 2 * x;
    pp[e] = 0.25f * (pb[base] + pb[base + 1] + pb[base + WW] + pb[base + WW + 1]);
    pc[e] = 0.25f * (cb[base] + cb[base + 1] + cb[base + WW] + cb[base + WW + 1]);
  }
  __syncthreads();

  const int nbase = y * PW;
  for (int e = t; e < PW * MIDCH; e += 256) {
    const int x = e >> 5, m = e & 31;
    float at = 0.f, ap = 0.f, ag = 0.f;
#pragma unroll 8
    for (int c = 0; c < INCH; ++c) {
      const float pcx = pc[c * PW + x];
      const float ppx = pp[c * PW + x];
      at = fmaf(wt[m * INCH + c], pcx, at);
      ap = fmaf(wp[m * INCH + c], ppx, ap);
      ag = fmaf(wg[m * INCH + c], ppx, ag);
    }
    const size_t n = (size_t)nbase + x;
    thetaB[((size_t)b * NTOK + n) * MIDCH + m] = (__bf16)at;
    phiB  [((size_t)b * NTOK + n) * MIDCH + m] = (__bf16)ap;
    gT    [(size_t)b * MIDCH * NTOK + (size_t)m * NTOK + n] = (__bf16)ag;
  }
}

// ---------------------------------------------------------------------------
// Kernel 2: flash attention.  64 queries/block (4 waves x 16 rows), 32-key
// tiles double-buffered in LDS via the Tensor Data Mover (wave 0 issues the
// DMA for tile k+1 while all waves compute on tile k).  Online softmax with
// DPP butterflies.  Writes y[b][mid][n] f32.
// grid = (NTOK/64, BATCH), block = 128
// ---------------------------------------------------------------------------
__global__ void attn_kernel(const __bf16* __restrict__ thetaB,
                            const __bf16* __restrict__ phiB,
                            const __bf16* __restrict__ gT,
                            float* __restrict__ y) {
  const int b     = blockIdx.y;
  const int qtile = blockIdx.x;
  const int lane  = threadIdx.x & 31;
  const int wave  = threadIdx.x >> 5;
  const bool hi   = lane >= 16;
  const int lrow  = hi ? lane - 16 : lane;
  const int qbase = qtile * 64 + wave * 16;

  __shared__ __bf16 sPhi[2][32 * 32];   // [key][mid]
  __shared__ __bf16 sG[2][32 * 32];     // [mid][key]
  __shared__ __bf16 sP[4][16 * 32];     // per-wave P tile [q][key]

  const __bf16* thb = thetaB + (size_t)b * NTOK * MIDCH;
  const __bf16* phb = phiB   + (size_t)b * NTOK * MIDCH;
  const __bf16* gb  = gT     + (size_t)b * MIDCH * NTOK;

  // ---- tile "issue" (DMA or cooperative fallback) -------------------------
  auto issue_tiles = [&](int buf, int k0) {
#if USE_TDM
    if (wave == 0) {
      // phi tile: 1024 contiguous bf16 (1-D tile)
      tdm_load_2d_bf16(&sPhi[buf][0], phb + (size_t)k0 * MIDCH,
                       1024u, 0u, (unsigned)NTOK * MIDCH, 1u,
                       (unsigned long long)NTOK * MIDCH);
      // gT tile: 32 rows x 32 keys, row stride NTOK elements (2-D tile)
      tdm_load_2d_bf16(&sG[buf][0], gb + k0,
                       32u, 32u, (unsigned)NTOK, (unsigned)MIDCH,
                       (unsigned long long)NTOK);
    }
#else
    {
      const unsigned int* src = (const unsigned int*)(phb + (size_t)k0 * MIDCH);
      unsigned int* dst = (unsigned int*)&sPhi[buf][0];
      for (int i = threadIdx.x; i < 512; i += 128) dst[i] = src[i];
      unsigned int* dg = (unsigned int*)&sG[buf][0];
      for (int i = threadIdx.x; i < 512; i += 128) {
        const int row = i >> 4, pr = i & 15;
        dg[row * 16 + pr] =
            *(const unsigned int*)(gb + (size_t)row * NTOK + k0 + 2 * pr);
      }
    }
#endif
  };

  // query A-fragment (row = this lane's query, K = mid, contiguous)
  const v16bf aQ = frag_from_row(thb + (size_t)(qbase + lrow) * MIDCH, hi);

  v8f o0 = {}, o1 = {};
  float mrow[8], lsum[8];
#pragma unroll
  for (int r = 0; r < 8; ++r) { mrow[r] = -__builtin_inff(); lsum[r] = 0.f; }

  issue_tiles(0, 0);   // prologue

  for (int k0 = 0, it = 0; k0 < NTOK; k0 += 32, ++it) {
    const int cur = it & 1;
#if USE_TDM
    if (wave == 0) __builtin_amdgcn_s_wait_tensorcnt(0);
#endif
    __syncthreads();   // current tiles visible; prev iter done with buf^1
    if (k0 + 32 < NTOK) issue_tiles(cur ^ 1, k0 + 32);

    const __bf16* tPhi = sPhi[cur];
    const __bf16* tG   = sG[cur];

    // S = theta x phi^T (two 16x16 key-column tiles)
    v8f s0 = {}, s1 = {};
    {
      const v16bf bK0 = frag_from_row(tPhi + lrow * 32, hi);
      const v16bf bK1 = frag_from_row(tPhi + (16 + lrow) * 32, hi);
      s0 = wmma_bf16(aQ, bK0, s0);
      s1 = wmma_bf16(aQ, bK1, s1);
    }

    // online softmax (row = one 16-lane half at VGPR index r)
    float scale[8];
#pragma unroll
    for (int r = 0; r < 8; ++r) {
      const float mx   = redmax16(fmaxf(s0[r], s1[r]));
      const float mnew = fmaxf(mrow[r], mx);
      const float sc   = __expf(mrow[r] - mnew);
      const float p0   = __expf(s0[r] - mnew);
      const float p1   = __expf(s1[r] - mnew);
      lsum[r] = lsum[r] * sc + redsum16(p0 + p1);
      mrow[r] = mnew;
      scale[r] = sc;
      s0[r] = p0; s1[r] = p1;
    }

    // re-layout P (C-layout -> A-layout) through per-wave LDS scratch
    __bf16* Pw = sP[wave];
#pragma unroll
    for (int r = 0; r < 8; ++r) {
      const int row = hi ? r + 8 : r;
      Pw[row * 32 + lrow]      = (__bf16)s0[r];
      Pw[row * 32 + 16 + lrow] = (__bf16)s1[r];
    }
    __syncthreads();   // P visible

#pragma unroll
    for (int r = 0; r < 8; ++r) { o0[r] *= scale[r]; o1[r] *= scale[r]; }

    const v16bf aP  = frag_from_row(Pw + lrow * 32, hi);
    const v16bf bG0 = frag_from_row(tG + lrow * 32, hi);
    const v16bf bG1 = frag_from_row(tG + (16 + lrow) * 32, hi);
    o0 = wmma_bf16(aP, bG0, o0);
    o1 = wmma_bf16(aP, bG1, o1);
  }

  float* yb = y + (size_t)b * MIDCH * NTOK;
#pragma unroll
  for (int r = 0; r < 8; ++r) {
    const int row = hi ? r + 8 : r;
    const float inv = 1.0f / lsum[r];
    yb[(size_t)lrow * NTOK + qbase + row]        = o0[r] * inv;
    yb[(size_t)(16 + lrow) * NTOK + qbase + row] = o1[r] * inv;
  }
}

// ---------------------------------------------------------------------------
// Kernel 3: conv weight transpose  w_out[oc][c][tap] -> wT[tap][oc][c] bf16
// ---------------------------------------------------------------------------
__global__ void wprep_kernel(const float* __restrict__ w_out,
                             __bf16* __restrict__ wT) {
  const int i = blockIdx.x * 256 + threadIdx.x;
  if (i < INCH * MIDCH * 9) {
    const int tap = i % 9;
    const int c   = (i / 9) % MIDCH;
    const int oc  = i / (9 * MIDCH);
    wT[((size_t)tap * INCH + oc) * MIDCH + c] = (__bf16)w_out[i];
  }
}

// ---------------------------------------------------------------------------
// Kernel 4: fused bilinear upsample (align_corners, x2) + 3x3 SAME conv as
// implicit GEMM (K = 32 ch per tap, 9 taps).  grid = (8,8,BATCH), block 256.
// ---------------------------------------------------------------------------
__global__ void conv_kernel(const float* __restrict__ y,
                            const __bf16* __restrict__ wT,
                            float* __restrict__ out) {
  const int b   = blockIdx.z;
  const int oy0 = blockIdx.y * 16;
  const int ox0 = blockIdx.x * 16;
  const int t   = threadIdx.x;

  __shared__ __bf16 patch[18 * 18 * MIDCH];
  __shared__ __bf16 sw[9 * INCH * MIDCH];

  {
    const unsigned int* src = (const unsigned int*)wT;
    unsigned int* dst = (unsigned int*)sw;
    for (int i = t; i < (9 * INCH * MIDCH) / 2; i += 256) dst[i] = src[i];
  }

  const float* yb = y + (size_t)b * MIDCH * NTOK;
  const float ratio = 63.0f / 127.0f;
  for (int e = t; e < 18 * 18 * MIDCH; e += 256) {
    const int c  = e & 31;
    const int px = (e >> 5) % 18;
    const int py = (e >> 5) / 18;
    const int gy = oy0 - 1 + py;
    const int gx = ox0 - 1 + px;
    float v = 0.f;
    if (gy >= 0 && gy < HH && gx >= 0 && gx < WW) {
      const float fy = gy * ratio, fx = gx * ratio;
      const int y0i = (int)fy, x0i = (int)fx;
      const int y1i = y0i + 1 < PH ? y0i + 1 : PH - 1;
      const int x1i = x0i + 1 < PW ? x0i + 1 : PW - 1;
      const float wy = fy - (float)y0i, wx = fx - (float)x0i;
      const float* yc = yb + (size_t)c * NTOK;
      const float v00 = yc[y0i * PW + x0i], v01 = yc[y0i * PW + x1i];
      const float v10 = yc[y1i * PW + x0i], v11 = yc[y1i * PW + x1i];
      v = (v00 * (1.f - wx) + v01 * wx) * (1.f - wy) +
          (v10 * (1.f - wx) + v11 * wx) * wy;
    }
    patch[e] = (__bf16)v;
  }
  __syncthreads();

  const int lane = t & 31, wave = t >> 5;
  const bool hi  = lane >= 16;
  const int lrow = hi ? lane - 16 : lane;

  for (int jj = 0; jj < 8; ++jj) {
    const int j   = wave * 8 + jj;
    const int ty  = j >> 2;
    const int oct = j & 3;
    v8f acc = {};
#pragma unroll
    for (int tap = 0; tap < 9; ++tap) {
      const int dy = tap / 3 - 1, dx = tap % 3 - 1;
      const v16bf a = frag_from_row(
          &patch[(((ty + dy + 1) * 18) + (lrow + dx + 1)) * MIDCH], hi);
      const v16bf w = frag_from_row(
          &sw[((size_t)tap * INCH + oct * 16 + lrow) * MIDCH], hi);
      acc = wmma_bf16(a, w, acc);
    }
    const int oc = oct * 16 + lrow;
#pragma unroll
    for (int r = 0; r < 8; ++r) {
      const int m = hi ? r + 8 : r;
      out[(((size_t)b * 2 * INCH + INCH + oc) * HH + (oy0 + ty)) * WW +
          ox0 + m] = acc[r];
    }
  }
}

// ---------------------------------------------------------------------------
// Kernel 5: concat channels 0..63 = primary (float4 copy)
// ---------------------------------------------------------------------------
__global__ void copy_kernel(const float* __restrict__ primary,
                            float* __restrict__ out) {
  const size_t i = (size_t)blockIdx.x * 256 + threadIdx.x;
  const size_t per_b = (size_t)INCH * HH * WW / 4;
  const size_t bb = i / per_b;
  const size_t r  = i - bb * per_b;
  ((float4*)out)[bb * 2 * per_b + r] = ((const float4*)primary)[i];
}

// ---------------------------------------------------------------------------
extern "C" void kernel_launch(void* const* d_in, const int* in_sizes, int n_in,
                              void* d_out, int out_size, void* d_ws,
                              size_t ws_size, hipStream_t stream) {
  const float* primary = (const float*)d_in[0];
  const float* cross   = (const float*)d_in[1];
  const float* w_theta = (const float*)d_in[2];
  const float* w_phi   = (const float*)d_in[3];
  const float* w_g     = (const float*)d_in[4];
  const float* w_out   = (const float*)d_in[5];
  float* out = (float*)d_out;

  char* ws = (char*)d_ws;
  __bf16* thetaB = (__bf16*)(ws);                 // 2 MiB
  __bf16* phiB   = (__bf16*)(ws + (2ull << 20));  // 2 MiB
  __bf16* gT     = (__bf16*)(ws + (4ull << 20));  // 2 MiB
  float*  yws    = (float*)(ws + (6ull << 20));   // 4 MiB
  __bf16* wT     = (__bf16*)(ws + (10ull << 20)); // 36 KiB

  prep_kernel<<<dim3(PH, BATCH), 256, 0, stream>>>(
      primary, cross, w_theta, w_phi, w_g, thetaB, phiB, gT);

  wprep_kernel<<<(INCH * MIDCH * 9 + 255) / 256, 256, 0, stream>>>(w_out, wT);

  attn_kernel<<<dim3(NTOK / 64, BATCH), 128, 0, stream>>>(thetaB, phiB, gT,
                                                          yws);

  conv_kernel<<<dim3(8, 8, BATCH), 256, 0, stream>>>(yws, wT, out);

  const size_t nf4 = (size_t)BATCH * INCH * HH * WW / 4;
  copy_kernel<<<(unsigned)(nf4 / 256), 256, 0, stream>>>(primary, out);
}
